// SEWYAttention_66889820668626
// MI455X (gfx1250) — matrix operand order repaired and checked by reading, not compiled, not run on hardware
//
#include <hip/hip_runtime.h>
#include <cstdint>

// ---------------------------------------------------------------------------
// Gemma2-style attention block for MI455X (gfx1250), wave32 + WMMA f16.
// Compute-bound (~185 GFLOP vs ~130MB traffic) -> v_wmma_f32_16x16x32_f16
// with f32 accumulation. Softcap bounds logits to (-50,50) so softmax is a
// single streaming pass (no running-max rescale needed).
// GEMMs stage the shared B panel into LDS with async global->LDS copies
// (ASYNCcnt path) so 8 waves share one 4KB panel per k-step.
// ---------------------------------------------------------------------------

typedef _Float16 v16h __attribute__((ext_vector_type(16)));
typedef _Float16 v8h  __attribute__((ext_vector_type(8)));
typedef float    v8f  __attribute__((ext_vector_type(8)));

union Frag { v16h v; v8h h[2]; };

__device__ __forceinline__ v8f wmma_f16(const Frag& a, const Frag& b, v8f c) {
  // (neg_a, A, neg_b, B, c_mod, C, reuse_a, reuse_b)
  return __builtin_amdgcn_wmma_f32_16x16x32_f16(false, a.v, false, b.v,
                                                (short)0, c, false, false);
}

constexpr int BB   = 2;
constexpr int SS   = 2048;
constexpr int HIDC = 2304;
constexpr int NHC  = 8;
constexpr int NKVC = 4;
constexpr int HDC  = 256;
constexpr int QDIM  = NHC * HDC;   // 2048
constexpr int KVDIM = NKVC * HDC;  // 1024
constexpr float SCALE_C   = 0.0625f;  // 256^-0.5
constexpr float SOFTCAP_C = 50.0f;

// ---------------- conversion / layout kernels ----------------

__global__ void cvt_f32_f16(const float* __restrict__ in, _Float16* __restrict__ out, int n) {
  int i = blockIdx.x * blockDim.x + threadIdx.x;
  if (i < n) out[i] = (_Float16)in[i];
}

// in[K][N] (row-major f32) -> out[N][K] (row-major f16)
__global__ void cvt_transpose(const float* __restrict__ in, _Float16* __restrict__ out,
                              int K, int N) {
  int i = blockIdx.x * blockDim.x + threadIdx.x;
  if (i < K * N) {
    int k = i / N, n = i % N;
    out[(size_t)n * K + k] = (_Float16)in[i];
  }
}

// Vh [B*S, NKV*HD] -> Vt [B, NKV, HD, S]  (so PV B-fragments are contiguous in key)
__global__ void vtrans(const _Float16* __restrict__ Vh, _Float16* __restrict__ Vt) {
  long long i = (long long)blockIdx.x * blockDim.x + threadIdx.x;
  const long long total = (long long)BB * SS * KVDIM;
  if (i >= total) return;
  int d  = (int)(i % HDC);  long long r = i / HDC;
  int hv = (int)(r % NKVC); r /= NKVC;
  int s  = (int)(r % SS);
  int b  = (int)(r / SS);
  Vt[(((size_t)b * NKVC + hv) * HDC + d) * SS + s] =
      Vh[(size_t)(b * SS + s) * KVDIM + hv * HDC + d];
}

// RoPE in place on Q [B,S,NH*HD] and K [B,S,NKV*HD] (f16)
__global__ void rope_inplace(_Float16* __restrict__ Qh, _Float16* __restrict__ Kh) {
  const int PH = HDC / 2;  // 128
  long long i = (long long)blockIdx.x * blockDim.x + threadIdx.x;
  const long long total = (long long)BB * SS * (NHC + NKVC) * PH;
  if (i >= total) return;
  int d = (int)(i % PH);  long long r = i / PH;
  int head = (int)(r % (NHC + NKVC)); r /= (NHC + NKVC);
  int s = (int)(r % SS);
  int b = (int)(r / SS);
  // inv_freq = 10000^{-2d/256} = exp(-ln(10000)*2d/256)
  float inv = __expf(-9.210340371976184f * (float)(2 * d) / (float)HDC);
  float ang = (float)s * inv;
  float sn, cs;
  __sincosf(ang, &sn, &cs);
  _Float16* p;
  if (head < NHC) p = Qh + (size_t)(b * SS + s) * QDIM  + head * HDC;
  else            p = Kh + (size_t)(b * SS + s) * KVDIM + (head - NHC) * HDC;
  float x1 = (float)p[d], x2 = (float)p[d + PH];
  p[d]      = (_Float16)(x1 * cs - x2 * sn);
  p[d + PH] = (_Float16)(x2 * cs + x1 * sn);
}

// ---------------- WMMA GEMM: C[M,N] = A[M,K] * Bt[N,K]^T ----------------
// Block = 256 threads = 8 waves; block tile 256x64. Each wave owns a 32x64
// tile (2 A-frags x 4 B-frags = 8 WMMAs per 32-k step). The 64x32 B panel is
// staged into LDS once per k-step via async global->LDS (all 8 waves share).

template <bool F32OUT>
__global__ __launch_bounds__(256)
void gemm256x64(const _Float16* __restrict__ A, const _Float16* __restrict__ Bt,
                void* __restrict__ Cout, int M, int N, int K) {
  __shared__ _Float16 Bpanel[64 * 32];  // 4KB per k-step
  const int lane = threadIdx.x & 31;
  const int widx = threadIdx.x >> 5;
  const int ntile = N >> 6;
  const int m0 = (blockIdx.x / ntile) * 256 + widx * 32;
  const int n0 = (blockIdx.x % ntile) * 64;
  const int c  = lane & 15;
  const int hi = lane >> 4;

  // Cooperative async staging: thread t moves 16B of B row (t>>2), k-off (t&3)*8.
  const int t = (int)threadIdx.x;
  const int srow  = t >> 2;
  const int skoff = (t & 3) * 8;
  const _Float16* sgp = Bt + (size_t)(n0 + srow) * K + skoff;
  const unsigned lds_dst =
      (unsigned)(uintptr_t)(&Bpanel[srow * 32 + skoff]);  // flat LDS addr low 32b

  v8f acc[2][4] = {};
  const _Float16* arow0 = A + (size_t)(m0 + c) * K;
  const _Float16* arow1 = A + (size_t)(m0 + 16 + c) * K;

  for (int kb = 0; kb < K; kb += 32) {
    // async global->LDS copy of the 64x32 B panel (one b128 per lane)
    unsigned long long gaddr = (unsigned long long)(uintptr_t)(sgp + kb);
    asm volatile("global_load_async_to_lds_b128 %0, %1, off"
                 :: "v"(lds_dst), "v"(gaddr) : "memory");
    asm volatile("s_wait_asynccnt 0" ::: "memory");
    __syncthreads();

    Frag a0, a1;
    a0.h[0] = *(const v8h*)(arow0 + kb + hi * 8);
    a0.h[1] = *(const v8h*)(arow0 + kb + 16 + hi * 8);
    a1.h[0] = *(const v8h*)(arow1 + kb + hi * 8);
    a1.h[1] = *(const v8h*)(arow1 + kb + 16 + hi * 8);
    #pragma unroll
    for (int nt = 0; nt < 4; ++nt) {
      Frag bf;
      const _Float16* bp = &Bpanel[(nt * 16 + c) * 32 + hi * 8];
      bf.h[0] = *(const v8h*)(bp);
      bf.h[1] = *(const v8h*)(bp + 16);
      acc[0][nt] = wmma_f16(a0, bf, acc[0][nt]);
      acc[1][nt] = wmma_f16(a1, bf, acc[1][nt]);
    }
    __syncthreads();  // protect panel before next iteration's async overwrite
  }

  #pragma unroll
  for (int rb = 0; rb < 2; ++rb) {
    #pragma unroll
    for (int r = 0; r < 8; ++r) {
      size_t row = (size_t)(m0 + rb * 16 + hi * 8 + r);
      #pragma unroll
      for (int nt = 0; nt < 4; ++nt) {
        if (F32OUT) {
          ((float*)Cout)[row * N + n0 + nt * 16 + c] = acc[rb][nt][r];
        } else {
          ((_Float16*)Cout)[row * N + n0 + nt * 16 + c] = (_Float16)acc[rb][nt][r];
        }
      }
    }
  }
}

// ---------------- fused causal GQA attention ----------------
// One wave = (batch b, head h, 16-query tile, half of head-dim 128 cols).
// Streaming softmax: softcap bounds logits to +-50 so exp() never overflows.

__global__ __launch_bounds__(256)
void attn_fused(const _Float16* __restrict__ Qh, const _Float16* __restrict__ Kh,
                const _Float16* __restrict__ Vt, _Float16* __restrict__ Ah) {
  __shared__ _Float16 lds_p[8][16 * 32];  // per-wave P-tile relayout scratch
  const int lane = threadIdx.x & 31;
  const int widx = threadIdx.x >> 5;
  const int wave = blockIdx.x * 8 + widx;
  const int QT = SS / 16;  // 128
  int t0 = wave;
  const int qtile = t0 % QT; t0 /= QT;
  const int dhalf = t0 & 1;  t0 >>= 1;
  const int h = t0 % NHC;    t0 /= NHC;
  const int b = t0;
  const int hkv = h >> 1;  // GROUPS = 2
  const int n  = lane & 15;
  const int hi = lane >> 4;

  // Q fragments for the whole 16x256 tile: 8 x (A 16x32) = 64 VGPRs
  Frag qf[8];
  const _Float16* qptr = Qh + (size_t)(b * SS + qtile * 16 + n) * QDIM + h * HDC;
  #pragma unroll
  for (int cc = 0; cc < 8; ++cc) {
    qf[cc].h[0] = *(const v8h*)(qptr + cc * 32 + hi * 16);
    qf[cc].h[1] = *(const v8h*)(qptr + cc * 32 + hi * 16 + 8);
  }

  v8f acc[8] = {};        // 16 x 128 output accumulator (this wave's d-half)
  float dsum[8] = {};     // per-lane partial softmax denominators

  const _Float16* kbase = Kh + (size_t)(b * SS) * KVDIM + hkv * HDC;
  const _Float16* vbase = Vt + ((size_t)(b * NKVC + hkv) * HDC + dhalf * 128) * SS;
  _Float16* pl = lds_p[widx];

  const int nk32 = qtile / 2 + 1;  // 32-key blocks needed under causal mask
  for (int kb32 = 0; kb32 < nk32; ++kb32) {
    const int key0 = kb32 * 32;

    // ---- S = Q . K^T for 32 keys: two 16x16 score tiles, 8 WMMAs each ----
    v8f s0 = {}, s1 = {};
    const _Float16* kr0 = kbase + (size_t)(key0 + n) * KVDIM;
    const _Float16* kr1 = kbase + (size_t)(key0 + 16 + n) * KVDIM;
    #pragma unroll
    for (int cc = 0; cc < 8; ++cc) {
      Frag k0f, k1f;
      k0f.h[0] = *(const v8h*)(kr0 + cc * 32 + hi * 16);
      k0f.h[1] = *(const v8h*)(kr0 + cc * 32 + hi * 16 + 8);
      k1f.h[0] = *(const v8h*)(kr1 + cc * 32 + hi * 16);
      k1f.h[1] = *(const v8h*)(kr1 + cc * 32 + hi * 16 + 8);
      s0 = wmma_f16(qf[cc], k0f, s0);
      s1 = wmma_f16(qf[cc], k1f, s1);
    }

    // ---- softcap + causal mask + exp; stage P into LDS (C->A relayout) ----
    #pragma unroll
    for (int r = 0; r < 8; ++r) {
      const int qrow = qtile * 16 + hi * 8 + r;
      float c0 = SOFTCAP_C * tanhf(s0[r] * (SCALE_C / SOFTCAP_C));
      float c1 = SOFTCAP_C * tanhf(s1[r] * (SCALE_C / SOFTCAP_C));
      float e0 = (key0 + n      <= qrow) ? __expf(c0) : 0.0f;
      float e1 = (key0 + 16 + n <= qrow) ? __expf(c1) : 0.0f;
      dsum[r] += e0 + e1;
      pl[(hi * 8 + r) * 32 + n]      = (_Float16)e0;
      pl[(hi * 8 + r) * 32 + 16 + n] = (_Float16)e1;
    }
    asm volatile("s_wait_dscnt 0" ::: "memory");

    Frag pf;  // P as A 16x32 fragment
    pf.h[0] = *(const v8h*)(pl + n * 32 + hi * 16);
    pf.h[1] = *(const v8h*)(pl + n * 32 + hi * 16 + 8);

    // ---- O += P . V : 8 WMMAs across the 128 output dims ----
    #pragma unroll
    for (int t = 0; t < 8; ++t) {
      Frag vf;
      const _Float16* vp = vbase + (size_t)(t * 16 + n) * SS + key0 + hi * 16;
      vf.h[0] = *(const v8h*)(vp);
      vf.h[1] = *(const v8h*)(vp + 8);
      acc[t] = wmma_f16(pf, vf, acc[t]);
    }
  }

  // Row sums: reduce partials across the 16 lanes of each half-wave.
  #pragma unroll
  for (int r = 0; r < 8; ++r) {
    float v = dsum[r];
    v += __shfl_xor(v, 1, 16);
    v += __shfl_xor(v, 2, 16);
    v += __shfl_xor(v, 4, 16);
    v += __shfl_xor(v, 8, 16);
    dsum[r] = v;
  }

  _Float16* aout = Ah + (size_t)(b * SS + qtile * 16 + hi * 8) * QDIM + h * HDC + dhalf * 128;
  #pragma unroll
  for (int t = 0; t < 8; ++t) {
    #pragma unroll
    for (int r = 0; r < 8; ++r) {
      aout[(size_t)r * QDIM + t * 16 + n] = (_Float16)(acc[t][r] / dsum[r]);
    }
  }
}

// ---------------- host launch ----------------

extern "C" void kernel_launch(void* const* d_in, const int* in_sizes, int n_in,
                              void* d_out, int out_size, void* d_ws, size_t ws_size,
                              hipStream_t stream) {
  const float* x  = (const float*)d_in[0];
  const float* wq = (const float*)d_in[1];
  const float* wk = (const float*)d_in[2];
  const float* wv = (const float*)d_in[3];
  const float* wo = (const float*)d_in[4];
  float* out = (float*)d_out;

  const size_t M = (size_t)BB * SS;  // 4096 total rows
  _Float16* ws = (_Float16*)d_ws;
  size_t off = 0;
  _Float16* Xh  = ws + off; off += M * HIDC;              // f16 hidden states
  _Float16* WqT = ws + off; off += (size_t)QDIM * HIDC;   // weights transposed [N,K]
  _Float16* WkT = ws + off; off += (size_t)KVDIM * HIDC;
  _Float16* WvT = ws + off; off += (size_t)KVDIM * HIDC;
  _Float16* WoT = ws + off; off += (size_t)HIDC * QDIM;
  _Float16* Qh  = ws + off; off += M * QDIM;
  _Float16* Kh  = ws + off; off += M * KVDIM;
  _Float16* Vh  = ws + off; off += M * KVDIM;
  _Float16* Vt  = Xh;   // alias: Xh dead after QKV GEMMs (stream-ordered)
  _Float16* Ah  = WqT;  // alias: Wq/Wk/Wv f16 copies dead after QKV GEMMs

  auto nblk = [](size_t n, int t) { return (unsigned)((n + (size_t)t - 1) / t); };

  cvt_f32_f16<<<nblk(M * HIDC, 256), 256, 0, stream>>>(x, Xh, (int)(M * HIDC));
  cvt_transpose<<<nblk((size_t)HIDC * QDIM, 256), 256, 0, stream>>>(wq, WqT, HIDC, QDIM);
  cvt_transpose<<<nblk((size_t)HIDC * KVDIM, 256), 256, 0, stream>>>(wk, WkT, HIDC, KVDIM);
  cvt_transpose<<<nblk((size_t)HIDC * KVDIM, 256), 256, 0, stream>>>(wv, WvT, HIDC, KVDIM);
  cvt_transpose<<<nblk((size_t)QDIM * HIDC, 256), 256, 0, stream>>>(wo, WoT, QDIM, HIDC);

  // QKV projections (f16 out); block tile 256x64
  gemm256x64<false><<<(unsigned)((M / 256) * (QDIM / 64)), 256, 0, stream>>>(
      Xh, WqT, Qh, (int)M, QDIM, HIDC);
  gemm256x64<false><<<(unsigned)((M / 256) * (KVDIM / 64)), 256, 0, stream>>>(
      Xh, WkT, Kh, (int)M, KVDIM, HIDC);
  gemm256x64<false><<<(unsigned)((M / 256) * (KVDIM / 64)), 256, 0, stream>>>(
      Xh, WvT, Vh, (int)M, KVDIM, HIDC);

  rope_inplace<<<nblk((size_t)BB * SS * (NHC + NKVC) * (HDC / 2), 256), 256, 0, stream>>>(Qh, Kh);
  vtrans<<<nblk(M * KVDIM, 256), 256, 0, stream>>>(Vh, Vt);

  // fused attention: B * NH * 2 d-halves * (S/16) q-tiles waves, 8 waves/block
  attn_fused<<<(unsigned)((size_t)BB * NHC * 2 * (SS / 16) / 8), 256, 0, stream>>>(
      Qh, Kh, Vt, Ah);

  // output projection (f32 out, straight into d_out)
  gemm256x64<true><<<(unsigned)((M / 256) * (HIDC / 64)), 256, 0, stream>>>(
      Ah, WoT, out, (int)M, HIDC, QDIM);
}
